// PrimVecQuantizedLSTM_16192026706036
// MI455X (gfx1250) — matrix operand order repaired and checked by reading, not compiled, not run on hardware
//
#include <hip/hip_runtime.h>
#include <math.h>

typedef __attribute__((ext_vector_type(16))) __bf16 v16bf;
typedef __attribute__((ext_vector_type(8)))  float  v8f;
typedef unsigned short u16;

// Dims: B=64, T=256, I=512, H=1024, 4H=4096, K=512 codes.

__device__ __forceinline__ u16 f32_to_bf16(float f) {
  unsigned int u = __float_as_uint(f);
  unsigned int r = u + 0x7FFFu + ((u >> 16) & 1u);   // round-to-nearest-even
  return (u16)(r >> 16);
}

// ---- Pack weights [Krows x 4096] f32 -> bf16 WMMA B-operand tiles ----------
// B tile (32x16): lane = nLocal + ((kk>>3)&1)<<4, slot = (kk&7)+((kk&16)>>1).
__global__ __launch_bounds__(256) void pack_w_kernel(const float* __restrict__ src,
                                                     u16* __restrict__ dst, int Krows) {
  int id = blockIdx.x * 256 + threadIdx.x;
  int k = id >> 12;            // /4096
  int n = id & 4095;
  if (k >= Krows) return;
  int kt = k >> 5, kk = k & 31;
  int nt = n >> 4, nL = n & 15;
  int lane = nL + (((kk >> 3) & 1) << 4);
  int slot = (kk & 7) + ((kk & 16) >> 1);
  dst[(((size_t)kt * 256 + nt) * 32 + lane) * 16 + slot] =
      f32_to_bf16(src[(size_t)k * 4096 + n]);
}

// ---- Pack sequence [64 x 256 x 512] f32 -> per-t bf16 WMMA A-operand tiles -
// A tile (16x32) interleave: lane = mL + ((kk>>3)&1)<<4, slot = (kk&7)+((kk&16)>>1).
__global__ __launch_bounds__(256) void pack_x_kernel(const float* __restrict__ seq,
                                                     u16* __restrict__ dst) {
  int id = blockIdx.x * 256 + threadIdx.x;   // < 64*256*512
  int m = id >> 17;
  int t = (id >> 9) & 255;
  int k = id & 511;
  int mt = m >> 4, mL = m & 15;
  int kt = k >> 5, kk = k & 31;
  int lane = mL + (((kk >> 3) & 1) << 4);
  int slot = (kk & 7) + ((kk & 16) >> 1);
  dst[(size_t)t * 32768 + (((size_t)mt * 16 + kt) * 32 + lane) * 16 + slot] =
      f32_to_bf16(seq[id]);
}

__global__ __launch_bounds__(256) void init_state_kernel(float* __restrict__ c,
                                                         u16* __restrict__ hP) {
  int id = blockIdx.x * 256 + threadIdx.x;   // 65536
  c[id]  = 0.0f;
  hP[id] = 0;
}

// ---- Fused recurrence step: GEMM (WMMA) + gates + state update -------------
// 64 WGs, one per 16-column hidden slab. WG computes the 4 gate N-tiles
// (n = g*1024 + [j0,j0+16)) for all 64 batch rows, exchanges z through LDS,
// then applies the LSTM cell for its slab. A = [x_t ; h] staged in LDS (192KB).
__global__ __launch_bounds__(256) void lstm_step_kernel(
    const u16* __restrict__ xP, const u16* __restrict__ hPin, u16* __restrict__ hPout,
    const u16* __restrict__ WxP, const u16* __restrict__ WhP,
    const float* __restrict__ bias, float* __restrict__ c,
    const int* __restrict__ n_iter, int t, float* __restrict__ unq) {
  extern __shared__ char smem[];
  u16*   As = (u16*)smem;                  // A tiles: [4 mt][48 kt] x 1KB = 192KB
  float* zs = (float*)(smem + 196608);     // z: [4 g][16 jl][64 m] f32 = 16KB

  const int tid  = threadIdx.x;
  const int lane = tid & 31;
  const int wave = tid >> 5;
  const int wg   = blockIdx.x;             // hidden slab 0..63

  // Stage A = [x_t ; h] into LDS: 12288 x 16B chunks, 48 per thread, no divergence.
  for (int cidx = tid; cidx < 12288; cidx += 256) {
    int tile = cidx >> 6;                  // 64 chunks per 1KB tile
    int cit  = cidx & 63;
    int mt = tile / 48, kt = tile - mt * 48;
    const u16* src = (kt < 16)
        ? (xP   + (size_t)(mt * 16 + kt)        * 512)
        : (hPin + (size_t)(mt * 32 + (kt - 16)) * 512);
    *(uint4*)(As + (size_t)tile * 512 + cit * 8) = *(const uint4*)(src + cit * 8);
  }
  __syncthreads();

  // Wave -> (gate, m-pair): 8 waves cover 4 gates x 4 M-tiles (2 tiles/wave).
  const int g   = wave >> 1;
  const int mt0 = (wave & 1) * 2;
  const int ntg = g * 64 + wg;             // global N tile for this gate/slab
  const int laneOff = lane * 16;           // u16 units (32B per lane)

  v8f acc0 = {0.f,0.f,0.f,0.f,0.f,0.f,0.f,0.f};
  v8f acc1 = {0.f,0.f,0.f,0.f,0.f,0.f,0.f,0.f};

  const u16* a0p = As + (size_t)(mt0 * 48)       * 512 + laneOff;
  const u16* a1p = As + (size_t)((mt0 + 1) * 48) * 512 + laneOff;
  const u16* bxp = WxP + (size_t)ntg * 512 + laneOff;
  const u16* bhp = WhP + (size_t)ntg * 512 + laneOff;

#pragma unroll 4
  for (int kt = 0; kt < 16; ++kt) {        // x part: K = 512
    v16bf a0 = *(const v16bf*)(a0p + (size_t)kt * 512);
    v16bf a1 = *(const v16bf*)(a1p + (size_t)kt * 512);
    v16bf bb = *(const v16bf*)(bxp + (size_t)kt * (256 * 512));
    acc0 = __builtin_amdgcn_wmma_f32_16x16x32_bf16(false, a0, false, bb, (short)0, acc0, false, false);
    acc1 = __builtin_amdgcn_wmma_f32_16x16x32_bf16(false, a1, false, bb, (short)0, acc1, false, false);
  }
#pragma unroll 4
  for (int kt = 0; kt < 32; ++kt) {        // h part: K = 1024
    v16bf a0 = *(const v16bf*)(a0p + (size_t)(16 + kt) * 512);
    v16bf a1 = *(const v16bf*)(a1p + (size_t)(16 + kt) * 512);
    v16bf bb = *(const v16bf*)(bhp + (size_t)kt * (256 * 512));
    acc0 = __builtin_amdgcn_wmma_f32_16x16x32_bf16(false, a0, false, bb, (short)0, acc0, false, false);
    acc1 = __builtin_amdgcn_wmma_f32_16x16x32_bf16(false, a1, false, bb, (short)0, acc1, false, false);
  }

  // Park z tiles in LDS: zs[(g*16 + jl)*64 + m], contiguous v8f per D fragment.
  {
    int jl = lane & 15;
    int hi = lane >> 4;
    float* zc = zs + ((size_t)g * 16 + jl) * 64;
    *(v8f*)(zc + mt0 * 16 + hi * 8)       = acc0;
    *(v8f*)(zc + (mt0 + 1) * 16 + hi * 8) = acc1;
  }
  __syncthreads();

  // Cell update: 64 rows x 16 cols = 1024 elements, 4 per thread.
  const int j0 = wg * 16;
#pragma unroll
  for (int kq = 0; kq < 4; ++kq) {
    int e  = tid + kq * 256;
    int m  = e & 63;
    int jl = e >> 6;
    int j  = j0 + jl;
    float zi = zs[(0 * 16 + jl) * 64 + m] + bias[j];
    float zf = zs[(1 * 16 + jl) * 64 + m] + bias[1024 + j];
    float zg = zs[(2 * 16 + jl) * 64 + m] + bias[2048 + j];
    float zo = zs[(3 * 16 + jl) * 64 + m] + bias[3072 + j];
    float ig = 1.0f / (1.0f + __expf(-zi));
    float fg = 1.0f / (1.0f + __expf(-zf));
    float gg = tanhf(zg);
    float og = 1.0f / (1.0f + __expf(-zo));
    size_t ci = (size_t)m * 1024 + j;
    float cn = fg * c[ci] + ig * gg;
    c[ci] = cn;
    float hn = og * tanhf(cn);
    // repack h into A-operand bf16 layout for the next step
    int mt = m >> 4, mL = m & 15;
    int kt = j >> 5, kk = j & 31;
    int ln = mL + (((kk >> 3) & 1) << 4);
    int slot = (kk & 7) + ((kk & 16) >> 1);
    hPout[(((size_t)mt * 32 + kt) * 32 + ln) * 16 + slot] = f32_to_bf16(hn);
    if (n_iter[m] == t) unq[ci] = hn;
  }
}

// ---- VQ: squared-L2 argmin over 512 codes, fp32, wave32 shuffle reduce -----
__global__ __launch_bounds__(256) void vq_kernel(const float* __restrict__ codebook,
                                                 float* __restrict__ out) {
  __shared__ float u[1024];
  __shared__ float wBest[8];
  __shared__ int   wIdx[8];
  __shared__ int   sBest;
  int b = blockIdx.x;
  int tid = threadIdx.x;
  const float* unq = out + 64 + 65536 + (size_t)b * 1024;
  for (int e = tid; e < 1024; e += 256) u[e] = unq[e];
  __syncthreads();
  int wave = tid >> 5, lane = tid & 31;
  float best = 3.4e38f; int bi = 0;
  for (int code = wave; code < 512; code += 8) {
    const float* cb = codebook + (size_t)code * 1024;
    float s = 0.0f;
    for (int e = lane; e < 1024; e += 32) { float d = u[e] - cb[e]; s += d * d; }
#pragma unroll
    for (int o = 16; o > 0; o >>= 1) s += __shfl_xor(s, o, 32);
    if (s < best) { best = s; bi = code; }
  }
  if (lane == 0) { wBest[wave] = best; wIdx[wave] = bi; }
  __syncthreads();
  if (tid == 0) {
    float bb = wBest[0]; int ii = wIdx[0];
    for (int w = 1; w < 8; ++w)
      if (wBest[w] < bb || (wBest[w] == bb && wIdx[w] < ii)) { bb = wBest[w]; ii = wIdx[w]; }
    sBest = ii;
    out[b] = (float)ii;
  }
  __syncthreads();
  int kbest = sBest;
  float* q = out + 64 + (size_t)b * 1024;
  for (int e = tid; e < 1024; e += 256) q[e] = codebook[(size_t)kbest * 1024 + e];
}

extern "C" void kernel_launch(void* const* d_in, const int* in_sizes, int n_in,
                              void* d_out, int out_size, void* d_ws, size_t ws_size,
                              hipStream_t stream) {
  (void)in_sizes; (void)n_in; (void)out_size; (void)ws_size;
  const float* seq      = (const float*)d_in[0];  // [64,256,512]
  const int*   n_iter   = (const int*)  d_in[1];  // [64]
  const float* Wx       = (const float*)d_in[2];  // [512,4096]
  const float* Wh       = (const float*)d_in[3];  // [1024,4096]
  const float* bias     = (const float*)d_in[4];  // [4096]
  const float* codebook = (const float*)d_in[5];  // [512,1024]
  float* out = (float*)d_out;   // [64 idx][64*1024 quantized][64*1024 unquantized]

  char* ws = (char*)d_ws;
  u16*   WxP = (u16*)(ws);                                       // 4 MB
  u16*   WhP = (u16*)(ws + ((size_t)4  << 20));                  // 8 MB
  u16*   xP  = (u16*)(ws + ((size_t)12 << 20));                  // 16 MB
  u16*   hP0 = (u16*)(ws + ((size_t)28 << 20));                  // 128 KB
  u16*   hP1 = (u16*)(ws + ((size_t)28 << 20) + (128u << 10));   // 128 KB
  float* c   = (float*)(ws + ((size_t)28 << 20) + (256u << 10)); // 256 KB

  pack_w_kernel<<<8192,  256, 0, stream>>>(Wx, WxP, 512);
  pack_w_kernel<<<16384, 256, 0, stream>>>(Wh, WhP, 1024);
  pack_x_kernel<<<32768, 256, 0, stream>>>(seq, xP);
  init_state_kernel<<<256, 256, 0, stream>>>(c, hP0);

  const size_t ldsBytes = 196608 + 16384;   // A stage (192KB) + z exchange (16KB)
  float* unq = out + 64 + 65536;
  for (int t = 0; t < 256; ++t) {
    const u16* hin = (t & 1) ? hP1 : hP0;
    u16*       hout = (t & 1) ? hP0 : hP1;
    lstm_step_kernel<<<64, 256, ldsBytes, stream>>>(
        xP + (size_t)t * 32768, hin, hout, WxP, WhP, bias, c, n_iter, t, unq);
  }
  vq_kernel<<<64, 256, 0, stream>>>(codebook, out);
}
